// SharedExpertPool_82626580841051
// MI455X (gfx1250) — matrix operand order repaired and checked by reading, not compiled, not run on hardware
//
#include <hip/hip_runtime.h>
#include <hip/hip_bf16.h>
#include <stdint.h>

#define T_TOK 2048
#define H_DIM 1024
#define D_DIM 512
#define E_NUM 64
#define K_TOP 2
#define ROWS_PAD (T_TOK * K_TOP + 64)
#define LSTR 40   // LDS row stride in bf16 elements: 80B, 16B-aligned, conflict-friendly

typedef __attribute__((ext_vector_type(16))) __bf16 v16bf;
typedef __attribute__((ext_vector_type(8)))  float  v8f;

union Frag { v16bf v; unsigned int u[8]; uint4 q[2]; };

__device__ __forceinline__ unsigned short bfbits(__bf16 h) {
    union { __bf16 h; unsigned short s; } c; c.h = h; return c.s;
}
__device__ __forceinline__ unsigned int pack2bf(float lo, float hi) {
    return (unsigned int)bfbits((__bf16)lo) | ((unsigned int)bfbits((__bf16)hi) << 16);
}

// ---------------- K0: zero output / usage / counts ----------------
__global__ void k_init(float* __restrict__ out, float* __restrict__ usage,
                       int* __restrict__ counts, long long n_out) {
    long long i0 = (long long)blockIdx.x * blockDim.x + threadIdx.x;
    long long stride = (long long)gridDim.x * blockDim.x;
    for (long long j = i0; j < n_out; j += stride) out[j] = 0.0f;
    if (i0 < E_NUM) { usage[i0] = 0.0f; counts[i0] = 0; }
}

// ---------------- K1: router (logits, top-2, usage, counts) ----------------
__global__ void k_router(const float* __restrict__ x, const float* __restrict__ gate,
                         int* __restrict__ topk_idx, float* __restrict__ topk_w,
                         float* __restrict__ usage, int* __restrict__ counts) {
    const int t = blockIdx.x;
    const int e = threadIdx.x;            // 64 threads, one expert each
    const float* xr = x + (long long)t * H_DIM;
    float acc = 0.0f;
    for (int h = 0; h < H_DIM; ++h) acc += xr[h] * gate[h * E_NUM + e];

    __shared__ float slog[E_NUM];
    __shared__ float s_m, s_d, s_l1, s_l2;
    __shared__ int   s_i1, s_i2;
    slog[e] = acc;
    __syncthreads();
    if (e == 0) {
        float l1 = -1e30f, l2 = -1e30f; int i1 = 0, i2 = 0;
        for (int j = 0; j < E_NUM; ++j) {
            float v = slog[j];
            if (v > l1)      { l2 = l1; i2 = i1; l1 = v; i1 = j; }
            else if (v > l2) { l2 = v;  i2 = j; }
        }
        float d = 0.0f;
        for (int j = 0; j < E_NUM; ++j) d += __expf(slog[j] - l1);
        s_m = l1; s_d = d; s_l1 = l1; s_l2 = l2; s_i1 = i1; s_i2 = i2;
    }
    __syncthreads();
    atomicAdd(&usage[e], __expf(acc - s_m) / s_d);   // routing_probs mean numerator
    if (e == 0) {
        float w0 = 1.0f / (1.0f + __expf(s_l2 - s_l1));   // softmax over top-2
        topk_idx[t * 2 + 0] = s_i1;  topk_idx[t * 2 + 1] = s_i2;
        topk_w [t * 2 + 0] = w0;     topk_w [t * 2 + 1] = 1.0f - w0;
        atomicAdd(&counts[s_i1], 1);
        atomicAdd(&counts[s_i2], 1);
    }
}

// ---------------- K2: load-balance loss ----------------
__global__ void k_lbloss(const float* __restrict__ usage, float* __restrict__ lb_out) {
    if (threadIdx.x == 0) {
        float s = 0.0f;
        for (int e2 = 0; e2 < E_NUM; ++e2) {
            float u = usage[e2] * (1.0f / (float)T_TOK);
            s += u * u;
        }
        *lb_out = (float)E_NUM * s;
    }
}

// ---------------- K3: prefix sum over expert counts ----------------
__global__ void k_offsets(const int* __restrict__ counts, int* __restrict__ offs,
                          int* __restrict__ cursor) {
    if (threadIdx.x == 0) {
        int acc = 0;
        for (int e2 = 0; e2 < E_NUM; ++e2) { offs[e2] = acc; cursor[e2] = acc; acc += counts[e2]; }
        offs[E_NUM] = acc;
    }
}

// ---------------- K4: scatter tokens into compact per-expert lists ----------------
__global__ void k_scatter(const int* __restrict__ topk_idx, const float* __restrict__ topk_w,
                          int* __restrict__ cursor, int* __restrict__ tok,
                          float* __restrict__ wgt) {
    int t = blockIdx.x * blockDim.x + threadIdx.x;
    if (t >= T_TOK) return;
    for (int k = 0; k < K_TOP; ++k) {
        int e = topk_idx[t * 2 + k];
        int pos = atomicAdd(&cursor[e], 1);
        tok[pos] = t;
        wgt[pos] = topk_w[t * 2 + k];
    }
}

// ---------------- K5: grouped up/gate GEMM + SiLU -> act (bf16, weight folded) ----------------
__global__ __launch_bounds__(256)
void k_expert_up(const float* __restrict__ x, const float* __restrict__ w1,
                 const float* __restrict__ w3, const int* __restrict__ offs,
                 const int* __restrict__ tok, const float* __restrict__ wgt,
                 __bf16* __restrict__ act) {
    const int e     = blockIdx.z;
    const int cnt   = offs[e + 1] - offs[e];
    const int mbase = blockIdx.y * 64;
    if (mbase >= cnt) return;
    const int rowbase = offs[e] + mbase;
    int rows = cnt - mbase; if (rows > 64) rows = 64;
    const int nbase = blockIdx.x * 64;

    __shared__ __bf16 xs [64][LSTR];   // row-major: [m][k]
    __shared__ __bf16 w1t[64][LSTR];   // transposed: [n][k]
    __shared__ __bf16 w3t[64][LSTR];   // transposed: [n][k]
    __shared__ int    tokS[64];
    __shared__ float  wgtS[64];

    const int tid = threadIdx.x;
    if (tid < 64) {
        tokS[tid] = tok[(tid < rows) ? (rowbase + tid) : rowbase];
        wgtS[tid] = (tid < rows) ? wgt[rowbase + tid] : 0.0f;
    }
    __syncthreads();

    const int lane  = tid & 31;
    const int wave  = tid >> 5;
    const int mtile = wave & 3;
    const int nt0   = (wave >> 2) * 2;

    v8f acc_h0 = {}, acc_h1 = {}, acc_g0 = {}, acc_g1 = {};

    const int xk2 = (tid & 15) * 2, xm = tid >> 4;   // x loader: 2 k per thread
    const int wn2 = (tid & 31) * 2, wk = tid >> 5;   // w loader: 2 n per thread

    for (int kc = 0; kc < H_DIM; kc += 32) {
        if (kc + 32 < H_DIM) {  // hint next weight chunk toward the caches
            __builtin_prefetch(&w1[((long long)e * H_DIM + kc + 32 + wk) * D_DIM + nbase + wn2], 0, 0);
            __builtin_prefetch(&w3[((long long)e * H_DIM + kc + 32 + wk) * D_DIM + nbase + wn2], 0, 0);
        }
        // x tile: 64m x 32k, row-major bf16, packed b32 stores
        #pragma unroll
        for (int i = 0; i < 4; ++i) {
            int m = xm + i * 16;
            float2 xv = *(const float2*)&x[(long long)tokS[m] * H_DIM + kc + xk2];
            *(unsigned int*)&xs[m][xk2] = pack2bf(xv.x, xv.y);
        }
        // w1/w3 tiles: 32k x 64n in memory -> transposed [n][k] in LDS
        #pragma unroll
        for (int i = 0; i < 4; ++i) {
            int k = wk + i * 8;
            long long base = ((long long)e * H_DIM + kc + k) * D_DIM + nbase + wn2;
            float2 a1 = *(const float2*)&w1[base];
            float2 a3 = *(const float2*)&w3[base];
            w1t[wn2][k]     = (__bf16)a1.x;  w1t[wn2 + 1][k] = (__bf16)a1.y;
            w3t[wn2][k]     = (__bf16)a3.x;  w3t[wn2 + 1][k] = (__bf16)a3.y;
        }
        __syncthreads();

        Frag a;
        const int arow = mtile * 16 + (lane & 15);
        const int khi  = (lane >> 4) * 8;
        a.q[0] = *(const uint4*)&xs[arow][khi];
        a.q[1] = *(const uint4*)&xs[arow][16 + khi];

        const int bn = lane & 15;
        const int kb = (lane >> 4) * 8;
        #pragma unroll
        for (int p = 0; p < 2; ++p) {
            const int brow = (nt0 + p) * 16 + bn;
            Frag b1, b3;
            b1.q[0] = *(const uint4*)&w1t[brow][kb];
            b1.q[1] = *(const uint4*)&w1t[brow][kb + 16];
            b3.q[0] = *(const uint4*)&w3t[brow][kb];
            b3.q[1] = *(const uint4*)&w3t[brow][kb + 16];
            if (p == 0) {
                acc_h0 = __builtin_amdgcn_wmma_f32_16x16x32_bf16(false, a.v, false, b1.v, (short)0, acc_h0, false, false);
                acc_g0 = __builtin_amdgcn_wmma_f32_16x16x32_bf16(false, a.v, false, b3.v, (short)0, acc_g0, false, false);
            } else {
                acc_h1 = __builtin_amdgcn_wmma_f32_16x16x32_bf16(false, a.v, false, b1.v, (short)0, acc_h1, false, false);
                acc_g1 = __builtin_amdgcn_wmma_f32_16x16x32_bf16(false, a.v, false, b3.v, (short)0, acc_g1, false, false);
            }
        }
        __syncthreads();
    }

    // epilogue: act = silu(h) * g * routing_weight, stored bf16
    #pragma unroll
    for (int p = 0; p < 2; ++p) {
        v8f hh = p ? acc_h1 : acc_h0;
        v8f gg = p ? acc_g1 : acc_g0;
        const int col = nbase + (nt0 + p) * 16 + (lane & 15);
        #pragma unroll
        for (int r = 0; r < 8; ++r) {
            int m = mtile * 16 + r + ((lane >> 4) ? 8 : 0);
            if (m < rows) {
                float h = hh[r], g = gg[r];
                float s = h / (1.0f + __expf(-h));
                act[(long long)(rowbase + m) * D_DIM + col] = (__bf16)(s * g * wgtS[m]);
            }
        }
    }
}

// ---------------- K6: grouped down GEMM, atomic scatter into out ----------------
__global__ __launch_bounds__(256)
void k_expert_down(const __bf16* __restrict__ act, const float* __restrict__ w2,
                   const int* __restrict__ offs, const int* __restrict__ tok,
                   float* __restrict__ out) {
    const int e     = blockIdx.z;
    const int cnt   = offs[e + 1] - offs[e];
    const int mbase = blockIdx.y * 64;
    if (mbase >= cnt) return;
    const int rowbase = offs[e] + mbase;
    int rows = cnt - mbase; if (rows > 64) rows = 64;
    const int nbase = blockIdx.x * 64;   // over H

    __shared__ __bf16 as_[64][LSTR];   // row-major: [m][k]
    __shared__ __bf16 w2t[64][LSTR];   // transposed: [n][k]
    __shared__ int    tokS[64];

    const int tid = threadIdx.x;
    if (tid < 64) tokS[tid] = tok[(tid < rows) ? (rowbase + tid) : rowbase];
    __syncthreads();

    const int lane  = tid & 31;
    const int wave  = tid >> 5;
    const int mtile = wave & 3;
    const int nt0   = (wave >> 2) * 2;

    v8f acc0 = {}, acc1 = {};

    const int am  = tid >> 2;          // async A copy: row 0..63
    const int akq = (tid & 3) * 8;     // k sub-chunk: 0,8,16,24 (16B each)
    const int wn2 = (tid & 31) * 2, wk = tid >> 5;

    // act row for async copy (invalid rows duplicate rowbase; masked in epilogue)
    const long long arow_g = (long long)((am < rows) ? (rowbase + am) : rowbase);

    for (int kc = 0; kc < D_DIM; kc += 32) {
        // A tile (64m x 32k bf16): async global->LDS, one B128 chunk per lane
        {
            unsigned int la = (unsigned int)(uintptr_t)&as_[am][akq];
            unsigned long long ga =
                (unsigned long long)(uintptr_t)&act[arow_g * D_DIM + kc + akq];
            asm volatile("global_load_async_to_lds_b128 %0, %1, off"
                         :: "v"(la), "v"(ga) : "memory");
        }
        // w2 tile: 32k x 64n -> transposed [n][k] bf16
        #pragma unroll
        for (int i = 0; i < 4; ++i) {
            int k = wk + i * 8;
            float2 a2 = *(const float2*)&w2[((long long)e * D_DIM + kc + k) * H_DIM + nbase + wn2];
            w2t[wn2][k]     = (__bf16)a2.x;
            w2t[wn2 + 1][k] = (__bf16)a2.y;
        }
        asm volatile("s_wait_asynccnt 0x0" ::: "memory");
        __syncthreads();

        Frag a;
        const int arow = mtile * 16 + (lane & 15);
        const int khi  = (lane >> 4) * 8;
        a.q[0] = *(const uint4*)&as_[arow][khi];
        a.q[1] = *(const uint4*)&as_[arow][16 + khi];

        const int bn = lane & 15;
        const int kb = (lane >> 4) * 8;
        #pragma unroll
        for (int p = 0; p < 2; ++p) {
            const int brow = (nt0 + p) * 16 + bn;
            Frag b;
            b.q[0] = *(const uint4*)&w2t[brow][kb];
            b.q[1] = *(const uint4*)&w2t[brow][kb + 16];
            if (p == 0)
                acc0 = __builtin_amdgcn_wmma_f32_16x16x32_bf16(false, a.v, false, b.v, (short)0, acc0, false, false);
            else
                acc1 = __builtin_amdgcn_wmma_f32_16x16x32_bf16(false, a.v, false, b.v, (short)0, acc1, false, false);
        }
        __syncthreads();
    }

    #pragma unroll
    for (int p = 0; p < 2; ++p) {
        v8f aa = p ? acc1 : acc0;
        const int col = nbase + (nt0 + p) * 16 + (lane & 15);
        #pragma unroll
        for (int r = 0; r < 8; ++r) {
            int m = mtile * 16 + r + ((lane >> 4) ? 8 : 0);
            if (m < rows)
                atomicAdd(&out[(long long)tokS[m] * H_DIM + col], aa[r]);
        }
    }
}

// ---------------- host side ----------------
extern "C" void kernel_launch(void* const* d_in, const int* in_sizes, int n_in,
                              void* d_out, int out_size, void* d_ws, size_t ws_size,
                              hipStream_t stream) {
    (void)in_sizes; (void)n_in; (void)out_size; (void)ws_size;
    const float* x    = (const float*)d_in[0];
    const float* w1   = (const float*)d_in[1];
    const float* w3   = (const float*)d_in[2];
    const float* w2   = (const float*)d_in[3];
    const float* gate = (const float*)d_in[4];
    float* out = (float*)d_out;                       // [T*H] out, then [1] lb_loss

    uint8_t* ws = (uint8_t*)d_ws;
    size_t o = 0;
    auto carve = [&](size_t bytes) -> void* {
        void* p = ws + o;
        o += (bytes + 255) & ~(size_t)255;
        return p;
    };
    int*    topk_idx = (int*)   carve(sizeof(int)   * T_TOK * K_TOP);
    float*  topk_w   = (float*) carve(sizeof(float) * T_TOK * K_TOP);
    float*  usage    = (float*) carve(sizeof(float) * E_NUM);
    int*    counts   = (int*)   carve(sizeof(int)   * E_NUM);
    int*    offs     = (int*)   carve(sizeof(int)   * (E_NUM + 1));
    int*    cursor   = (int*)   carve(sizeof(int)   * E_NUM);
    int*    tokc     = (int*)   carve(sizeof(int)   * ROWS_PAD);
    float*  wgtc     = (float*) carve(sizeof(float) * ROWS_PAD);
    __bf16* act      = (__bf16*)carve(sizeof(__bf16) * (size_t)ROWS_PAD * D_DIM);

    const long long n_out = (long long)T_TOK * H_DIM + 1;
    k_init   <<<2048, 256, 0, stream>>>(out, usage, counts, n_out);
    k_router <<<T_TOK, E_NUM, 0, stream>>>(x, gate, topk_idx, topk_w, usage, counts);
    k_lbloss <<<1, 64, 0, stream>>>(usage, out + (long long)T_TOK * H_DIM);
    k_offsets<<<1, 64, 0, stream>>>(counts, offs, cursor);
    k_scatter<<<T_TOK / 256, 256, 0, stream>>>(topk_idx, topk_w, cursor, tokc, wgtc);
    k_expert_up  <<<dim3(D_DIM / 64, T_TOK / 64, E_NUM), 256, 0, stream>>>(x, w1, w3, offs, tokc, wgtc, act);
    k_expert_down<<<dim3(H_DIM / 64, T_TOK / 64, E_NUM), 256, 0, stream>>>(act, w2, offs, tokc, out);
}
